// Block_26628797235524
// MI455X (gfx1250) — matrix operand order repaired and checked by reading, not compiled, not run on hardware
//
#include <hip/hip_runtime.h>
#include <hip/hip_bf16.h>

typedef _Float16 half16 __attribute__((ext_vector_type(16)));
typedef float    float8 __attribute__((ext_vector_type(8)));

#define B_   8
#define L_   4096
#define DIM_ 128
#define DIN_ 256
#define DST_ 16
#define DTR_ 8
#define KG_  2
#define CH_  64   // chunks
#define CL_  64   // chunk length

__device__ __forceinline__ float siluf(float v)     { return v / (1.f + __expf(-v)); }
__device__ __forceinline__ float softplusf(float v) { return v > 20.f ? v : __logf(1.f + __expf(v)); }

// A-matrix (16xK f16) lane layout as this code packs it (contiguous k, B matches):
// lanes 0-15 -> M rows, half=lane/16 selects K groups:
// elements 0..7  : K = k0 + half*16 + j
// elements 8..15 : K = k0 + half*16 + 8 + j
__device__ __forceinline__ half16 load_a_tile(const float* __restrict__ base, int stride,
                                              int mrow, int k0, int half_sel) {
  const float* p0 = base + (size_t)mrow * stride + k0 + half_sel * 16;
  const float* p1 = p0 + 8;
  half16 a;
#pragma unroll
  for (int j = 0; j < 8; ++j) { a[j] = (_Float16)p0[j]; a[8 + j] = (_Float16)p1[j]; }
  return a;
}

// B-matrix (Kx16 f16): lane n holds column n; elements 0..15 : K = k0 + half*16 + j
__device__ __forceinline__ half16 load_b_tile(const float* __restrict__ base, int stride,
                                              int ncol, int k0, int half_sel) {
  const float* p = base + (size_t)ncol * stride + k0 + half_sel * 16;
  half16 b;
#pragma unroll
  for (int j = 0; j < 16; ++j) b[j] = (_Float16)p[j];
  return b;
}

// ---------------- K1: in_proj GEMM + silu/conv-affine + scatter to scan layouts ----
// x:(B*L,128) @ Wi^T (512x128) -> 512 outs; o<256 -> xh path to XS(B,K,L,D); o>=256 -> z to ZB
__global__ void k1_inproj(const float* __restrict__ x, const float* __restrict__ Wi,
                          const float* __restrict__ cw, const float* __restrict__ cb,
                          float* __restrict__ XS, float* __restrict__ ZB) {
  int wave = (blockIdx.x * blockDim.x + threadIdx.x) >> 5;   // 65536 tiles
  int lane = threadIdx.x & 31;
  int mt = wave >> 5, nt = wave & 31;
  int m_base = mt * 16, n_base = nt * 16;
  int half = lane >> 4, lm = lane & 15;
  float8 c = {};
#pragma unroll
  for (int kc = 0; kc < 4; ++kc) {
    half16 a = load_a_tile(x,  DIM_, m_base + lm, kc * 32, half);
    half16 b = load_b_tile(Wi, DIM_, n_base + lm, kc * 32, half);
    c = __builtin_amdgcn_wmma_f32_16x16x32_f16(false, a, false, b, (short)0, c, false, false);
  }
  int o = n_base + lm;
#pragma unroll
  for (int r = 0; r < 8; ++r) {
    int m = m_base + r + half * 8;              // C layout: VGPR r -> M=r (+8 for hi lanes)
    int b_ = m >> 12, lthw = m & (L_ - 1);
    float v = c[r];
    if (o < DIN_) {
      v = siluf(v * cw[o] + cb[o]);
      int t = lthw >> 8, hh = (lthw >> 4) & 15, ww = lthw & 15;
      int lspe = ((hh * 16 + ww) << 4) + t;
      XS[(((size_t)(b_ * KG_ + 0) * L_) + lspe) * DIN_ + o] = v;                 // spectral order
      XS[(((size_t)(b_ * KG_ + 1) * L_) + (L_ - 1 - lthw)) * DIN_ + o] = v;      // flipped thw
    } else {
      ZB[((size_t)b_ * L_ + lthw) * DIN_ + (o - DIN_)] = siluf(v);
    }
  }
}

// ---------------- K2: x_dbl = x_proj_w[k](40x256) @ xs[b,k]^T -> dts/Bs/Cs ----------
__global__ void k2_xdbl(const float* __restrict__ XS, const float* __restrict__ xpw,
                        float* __restrict__ DTS, float* __restrict__ BS, float* __restrict__ CS) {
  int wave = (blockIdx.x * blockDim.x + threadIdx.x) >> 5;   // 16*3*256 = 12288 tiles
  int lane = threadIdx.x & 31;
  int bk = wave / (3 * 256);
  int rem = wave % (3 * 256);
  int ct = rem >> 8, lt = rem & 255;
  int k = bk & 1;
  const float* A  = xpw + (size_t)k * 40 * DIN_;
  const float* Bb = XS + (size_t)bk * L_ * DIN_;
  int lbase = lt * 16;
  int half = lane >> 4, lm = lane & 15;
  int crow = ct * 16 + lm;
  int arow = (crow < 40) ? crow : 0;
  float8 c = {};
#pragma unroll
  for (int kc = 0; kc < 8; ++kc) {
    half16 a = load_a_tile(A,  DIN_, arow,       kc * 32, half);
    half16 b = load_b_tile(Bb, DIN_, lbase + lm, kc * 32, half);
    c = __builtin_amdgcn_wmma_f32_16x16x32_f16(false, a, false, b, (short)0, c, false, false);
  }
  int l = lbase + lm;
#pragma unroll
  for (int r = 0; r < 8; ++r) {
    int cg = ct * 16 + r + half * 8;
    float v = c[r];
    if (cg < DTR_)            DTS[((size_t)bk * L_ + l) * DTR_ + cg] = v;
    else if (cg < DTR_ + DST_) BS[((size_t)bk * DST_ + (cg - DTR_)) * L_ + l] = v;
    else if (cg < 40)          CS[((size_t)bk * DST_ + (cg - DTR_ - DST_)) * L_ + l] = v;
  }
}

// ---------------- Pass 1: chunk-local scan (h from 0), record h_end and sum(delta) --
__global__ void p1_scan(const float* __restrict__ XS, const float* __restrict__ DTS,
                        const float* __restrict__ BS, const float* __restrict__ dtw,
                        const float* __restrict__ dtb, const float* __restrict__ alg,
                        float* __restrict__ HC, float* __restrict__ SD) {
  __shared__ float sB[DST_ * CL_];
  __shared__ float sDT[CL_ * DTR_];
  int blk = blockIdx.x;                 // B*K*CH = 1024
  int chunk = blk & (CH_ - 1), bk = blk >> 6;
  int k = bk & 1, d = threadIdx.x;
  int l0 = chunk * CL_;
  for (int i = threadIdx.x; i < DST_ * CL_; i += 256) {
    int n = i >> 6, s = i & (CL_ - 1);
    sB[n * CL_ + s] = BS[((size_t)bk * DST_ + n) * L_ + l0 + s];
  }
  {
    const float2* p = (const float2*)(DTS + ((size_t)bk * L_ + l0) * DTR_);
    ((float2*)sDT)[threadIdx.x] = p[threadIdx.x];      // 512 floats
  }
  __syncthreads();
  float w[DTR_];
#pragma unroll
  for (int r = 0; r < DTR_; ++r) w[r] = dtw[((size_t)(k * DIN_ + d)) * DTR_ + r];
  float bdt = dtb[k * DIN_ + d];
  float aA[DST_];
#pragma unroll
  for (int n = 0; n < DST_; ++n) aA[n] = -__expf(alg[((size_t)(k * DIN_ + d)) * DST_ + n]);
  float h[DST_];
#pragma unroll
  for (int n = 0; n < DST_; ++n) h[n] = 0.f;
  float sumD = 0.f;
  const float* xu = XS + ((size_t)bk * L_ + l0) * DIN_ + d;
  __builtin_prefetch(xu, 0, 1);
  for (int s = 0; s < CL_; ++s) {
    float dv = bdt;
#pragma unroll
    for (int r = 0; r < DTR_; ++r) dv += sDT[s * DTR_ + r] * w[r];
    float delta = softplusf(dv);
    sumD += delta;
    float du = delta * xu[s * DIN_];
#pragma unroll
    for (int n = 0; n < DST_; ++n)
      h[n] = __expf(delta * aA[n]) * h[n] + du * sB[n * CL_ + s];
  }
  size_t base = ((size_t)(bk * DIN_ + d) * CH_ + chunk) * DST_;
#pragma unroll
  for (int n = 0; n < DST_; ++n) HC[base + n] = h[n];
  SD[(size_t)(bk * DIN_ + d) * CH_ + chunk] = sumD;
}

// ---------------- Pass 2: carry propagation across chunks (per b,k,d,n) -------------
__global__ void p2_carry(const float* __restrict__ HC, const float* __restrict__ SD,
                         const float* __restrict__ alg, float* __restrict__ HS) {
  int tid = blockIdx.x * blockDim.x + threadIdx.x;   // 65536
  int n = tid & (DST_ - 1);
  int bkd = tid >> 4;
  int d = bkd & (DIN_ - 1), k = (bkd >> 8) & 1;
  float a = -__expf(alg[((size_t)(k * DIN_ + d)) * DST_ + n]);
  float hs = 0.f;
  for (int c = 0; c < CH_; ++c) {
    HS[((size_t)bkd * CH_ + c) * DST_ + n] = hs;
    float P = __expf(SD[(size_t)bkd * CH_ + c] * a);   // prod of exp(delta*a) over chunk
    hs = P * hs + HC[((size_t)bkd * CH_ + c) * DST_ + n];
  }
}

// ---------------- Pass 3: replay chunks with correct h0, emit y ---------------------
__global__ void p3_scan(const float* __restrict__ XS, const float* __restrict__ DTS,
                        const float* __restrict__ BS, const float* __restrict__ CS,
                        const float* __restrict__ dtw, const float* __restrict__ dtb,
                        const float* __restrict__ alg, const float* __restrict__ Dsv,
                        const float* __restrict__ HS, float* __restrict__ YB) {
  __shared__ float sB[DST_ * CL_];
  __shared__ float sC[DST_ * CL_];
  __shared__ float sDT[CL_ * DTR_];
  int blk = blockIdx.x;
  int chunk = blk & (CH_ - 1), bk = blk >> 6;
  int k = bk & 1, d = threadIdx.x;
  int l0 = chunk * CL_;
  for (int i = threadIdx.x; i < DST_ * CL_; i += 256) {
    int n = i >> 6, s = i & (CL_ - 1);
    sB[n * CL_ + s] = BS[((size_t)bk * DST_ + n) * L_ + l0 + s];
    sC[n * CL_ + s] = CS[((size_t)bk * DST_ + n) * L_ + l0 + s];
  }
  {
    const float2* p = (const float2*)(DTS + ((size_t)bk * L_ + l0) * DTR_);
    ((float2*)sDT)[threadIdx.x] = p[threadIdx.x];
  }
  __syncthreads();
  float w[DTR_];
#pragma unroll
  for (int r = 0; r < DTR_; ++r) w[r] = dtw[((size_t)(k * DIN_ + d)) * DTR_ + r];
  float bdt = dtb[k * DIN_ + d];
  float ds_ = Dsv[k * DIN_ + d];
  float aA[DST_];
#pragma unroll
  for (int n = 0; n < DST_; ++n) aA[n] = -__expf(alg[((size_t)(k * DIN_ + d)) * DST_ + n]);
  float h[DST_];
  size_t hbase = ((size_t)(bk * DIN_ + d) * CH_ + chunk) * DST_;
#pragma unroll
  for (int n = 0; n < DST_; ++n) h[n] = HS[hbase + n];
  const float* xu = XS + ((size_t)bk * L_ + l0) * DIN_ + d;
  float* yo = YB + ((size_t)bk * L_ + l0) * DIN_ + d;
  __builtin_prefetch(xu, 0, 1);
  for (int s = 0; s < CL_; ++s) {
    float dv = bdt;
#pragma unroll
    for (int r = 0; r < DTR_; ++r) dv += sDT[s * DTR_ + r] * w[r];
    float delta = softplusf(dv);
    float u = xu[s * DIN_];
    float du = delta * u;
    float y = 0.f;
#pragma unroll
    for (int n = 0; n < DST_; ++n) {
      h[n] = __expf(delta * aA[n]) * h[n] + du * sB[n * CL_ + s];
      y += h[n] * sC[n * CL_ + s];
    }
    yo[s * DIN_] = y + ds_ * u;
  }
}

// ---------------- K4a: combine fwd/rvs, LayerNorm over DIN, gate with z -------------
__global__ void k4_ln(const float* __restrict__ YB, const float* __restrict__ ZB,
                      const float* __restrict__ lnw, const float* __restrict__ lnb,
                      float* __restrict__ GB) {
  __shared__ float red[256];
  int bl = blockIdx.x;                  // B*L
  int b = bl >> 12, lthw = bl & (L_ - 1);
  int d = threadIdx.x;
  int t = lthw >> 8, hh = (lthw >> 4) & 15, ww = lthw & 15;
  int lspe = ((hh * 16 + ww) << 4) + t;
  float y = YB[(((size_t)(b * KG_ + 0) * L_) + lspe) * DIN_ + d]
          + YB[(((size_t)(b * KG_ + 1) * L_) + (L_ - 1 - lthw)) * DIN_ + d];
  red[d] = y; __syncthreads();
  for (int s = 128; s > 0; s >>= 1) { if (d < s) red[d] += red[d + s]; __syncthreads(); }
  float mu = red[0] * (1.f / DIN_); __syncthreads();
  float c0 = y - mu;
  red[d] = c0 * c0; __syncthreads();
  for (int s = 128; s > 0; s >>= 1) { if (d < s) red[d] += red[d + s]; __syncthreads(); }
  float rstd = rsqrtf(red[0] * (1.f / DIN_) + 1e-5f);
  float g = (c0 * rstd * lnw[d] + lnb[d]) * ZB[(size_t)bl * DIN_ + d];
  GB[(size_t)bl * DIN_ + d] = g;
}

// ---------------- K4b: out_proj GEMM (32768x256)@(256x128) via WMMA -----------------
__global__ void k5_outproj(const float* __restrict__ GB, const float* __restrict__ Wo,
                           float* __restrict__ out) {
  int wave = (blockIdx.x * blockDim.x + threadIdx.x) >> 5;   // 2048*8 tiles
  int lane = threadIdx.x & 31;
  int mt = wave >> 3, nt = wave & 7;
  int m_base = mt * 16, n_base = nt * 16;
  int half = lane >> 4, lm = lane & 15;
  float8 c = {};
#pragma unroll
  for (int kc = 0; kc < 8; ++kc) {
    half16 a = load_a_tile(GB, DIN_, m_base + lm, kc * 32, half);
    half16 b = load_b_tile(Wo, DIN_, n_base + lm, kc * 32, half);
    c = __builtin_amdgcn_wmma_f32_16x16x32_f16(false, a, false, b, (short)0, c, false, false);
  }
#pragma unroll
  for (int r = 0; r < 8; ++r) {
    int m = m_base + r + half * 8;
    out[(size_t)m * DIM_ + n_base + lm] = c[r];
  }
}

extern "C" void kernel_launch(void* const* d_in, const int* in_sizes, int n_in,
                              void* d_out, int out_size, void* d_ws, size_t ws_size,
                              hipStream_t stream) {
  const float* x   = (const float*)d_in[0];
  const float* Wi  = (const float*)d_in[1];
  const float* cw  = (const float*)d_in[2];
  const float* cb  = (const float*)d_in[3];
  const float* xpw = (const float*)d_in[4];
  const float* dtw = (const float*)d_in[5];
  const float* dtb = (const float*)d_in[6];
  const float* alg = (const float*)d_in[7];
  const float* Dsv = (const float*)d_in[8];
  const float* lnw = (const float*)d_in[9];
  const float* lnb = (const float*)d_in[10];
  const float* Wo  = (const float*)d_in[11];
  float* out = (float*)d_out;

  float* ws = (float*)d_ws;
  const size_t N_XS  = (size_t)B_ * KG_ * L_ * DIN_;       // 16.78M
  const size_t N_ZB  = (size_t)B_ * L_ * DIN_;             //  8.39M
  const size_t N_DTS = (size_t)B_ * KG_ * L_ * DTR_;       //  0.52M
  const size_t N_BS  = (size_t)B_ * KG_ * DST_ * L_;       //  1.05M
  const size_t N_HC  = (size_t)B_ * KG_ * DIN_ * CH_ * DST_; // 4.19M
  const size_t N_SD  = (size_t)B_ * KG_ * DIN_ * CH_;      //  0.26M
  float* XS  = ws;
  float* ZB  = XS  + N_XS;
  float* DTS = ZB  + N_ZB;
  float* BS  = DTS + N_DTS;
  float* CS  = BS  + N_BS;
  float* HC  = CS  + N_BS;
  float* SD  = HC  + N_HC;
  float* HS  = SD  + N_SD;
  float* YB  = HS  + N_HC;
  float* GB  = XS;   // XS dead after pass 3; reuse for gated-LN output

  k1_inproj <<<8192, 256, 0, stream>>>(x, Wi, cw, cb, XS, ZB);
  k2_xdbl   <<<1536, 256, 0, stream>>>(XS, xpw, DTS, BS, CS);
  p1_scan   <<<B_ * KG_ * CH_, 256, 0, stream>>>(XS, DTS, BS, dtw, dtb, alg, HC, SD);
  p2_carry  <<<256, 256, 0, stream>>>(HC, SD, alg, HS);
  p3_scan   <<<B_ * KG_ * CH_, 256, 0, stream>>>(XS, DTS, BS, CS, dtw, dtb, alg, Dsv, HS, YB);
  k4_ln     <<<B_ * L_, 256, 0, stream>>>(YB, ZB, lnw, lnb, GB);
  k5_outproj<<<2048, 256, 0, stream>>>(GB, Wo, out);
}
